// MultiGameDecisionTransformer_3470333575626
// MI455X (gfx1250) — matrix operand order repaired and checked by reading, not compiled, not run on hardware
//
#include <hip/hip_runtime.h>
#include <cstdint>
#include <cstddef>

// Problem constants (from reference)
#define B_  32
#define S_  512
#define H_  512
#define NH_ 8
#define NL_ 6
#define G_  8
#define D_  256
#define A_  18
#define T_  1024
#define HD_ 64
#define BS_ (B_ * S_)

typedef __bf16 bf16_t;
typedef __bf16 bf16x8  __attribute__((ext_vector_type(8)));
typedef __bf16 bf16x16 __attribute__((ext_vector_type(16)));
typedef float  f32x8   __attribute__((ext_vector_type(8)));
typedef int    v4i     __attribute__((ext_vector_type(4)));

union BF16Frag {
  bf16x16 v;
  bf16x8  h[2];
};

__device__ __forceinline__ bf16_t f2bf(float f) {
  unsigned u = __builtin_bit_cast(unsigned, f);
  unsigned r = u + 0x7FFFu + ((u >> 16) & 1u);   // round-to-nearest-even
  unsigned short hs = (unsigned short)(r >> 16);
  return __builtin_bit_cast(bf16_t, hs);
}

__device__ __forceinline__ f32x8 zero8() {
  f32x8 z = {0.f, 0.f, 0.f, 0.f, 0.f, 0.f, 0.f, 0.f};
  return z;
}

// ---------------------------------------------------------------------------
// gfx1250 async global->LDS copy (ASYNCcnt path), with sync fallback
// Builtin signature (from hipcc diagnostic): arg0 = v4i addrspace(1)* (global
// src), arg1 = v4i addrspace(3)* (LDS dst), arg2 = imm offset, arg3 = cpol.
// ---------------------------------------------------------------------------
#if defined(__has_builtin)
#if __has_builtin(__builtin_amdgcn_global_load_async_to_lds_b128)
#define HAS_ASYNC_LDS 1
#endif
#endif
#ifndef HAS_ASYNC_LDS
#define HAS_ASYNC_LDS 0
#endif

#if HAS_ASYNC_LDS
typedef __attribute__((address_space(1))) v4i* gas_ptr;
typedef __attribute__((address_space(3))) v4i* las_ptr;
__device__ __forceinline__ void async_b128(const void* g, void* l) {
  void* gnc = const_cast<void*>(g);
  __builtin_amdgcn_global_load_async_to_lds_b128((gas_ptr)gnc, (las_ptr)l, 0, 0);
}
#endif

__device__ __forceinline__ void copy_b128(const bf16_t* g, bf16_t* l) {
#if HAS_ASYNC_LDS
  async_b128((const void*)g, (void*)l);
#else
  *(bf16x8*)l = *(const bf16x8*)g;
#endif
}

// wait until at most one newer 4-op async group remains in flight
__device__ __forceinline__ void wait_async_4() {
#if HAS_ASYNC_LDS
#if __has_builtin(__builtin_amdgcn_s_wait_asynccnt)
  __builtin_amdgcn_s_wait_asynccnt(4);
#else
  asm volatile("s_wait_asynccnt 0x4" ::: "memory");
#endif
#endif
}

// wait until all async groups landed
__device__ __forceinline__ void wait_async_0() {
#if HAS_ASYNC_LDS
#if __has_builtin(__builtin_amdgcn_s_wait_asynccnt)
  __builtin_amdgcn_s_wait_asynccnt(0);
#else
  asm volatile("s_wait_asynccnt 0x0" ::: "memory");
#endif
#endif
}

// ---------------------------------------------------------------------------
// f32 -> bf16 flat conversion
// ---------------------------------------------------------------------------
__global__ void cvt_bf16_k(const float* __restrict__ src, bf16_t* __restrict__ dst, size_t n) {
  size_t stride = (size_t)gridDim.x * blockDim.x;
  for (size_t i = (size_t)blockIdx.x * blockDim.x + threadIdx.x; i < n; i += stride)
    dst[i] = f2bf(src[i]);
}

// f32 (G,K,N) -> bf16 (G,N,K)   (per-game weight transpose so WMMA B-frags
// become contiguous per-lane rows)
__global__ void cvt_bf16_t_k(const float* __restrict__ src, bf16_t* __restrict__ dst,
                             int G, int K, int N) {
  size_t total = (size_t)G * K * N;
  size_t stride = (size_t)gridDim.x * blockDim.x;
  for (size_t i = (size_t)blockIdx.x * blockDim.x + threadIdx.x; i < total; i += stride) {
    size_t g   = i / ((size_t)K * N);
    size_t rem = i - g * (size_t)K * N;
    int kk = (int)(rem / N);
    int nn = (int)(rem - (size_t)kk * N);
    dst[g * (size_t)K * N + (size_t)nn * K + kk] = f2bf(src[i]);
  }
}

// ---------------------------------------------------------------------------
// WMMA bf16 GEMM with async-LDS double buffering:
//   C[M,N] = A[M,K] * Bt[N,K]^T (+bias) (opt relu)
// Block tile 128x128 (8 waves, 4(M) x 2(N)); wave tile 32x64 (2x4 frags).
// K-step 32: stage A(128x32) + B(128x32) into LDS via
// GLOBAL_LOAD_ASYNC_TO_LDS_B128 (2-deep pipeline, s_wait_asynccnt), feed WMMA
// from LDS (rows padded to 40 elems -> conflict-free 16B ds reads).
// Final k-step peeled so the steady-state loop has a constant
// s_wait_asynccnt 4 (no per-iteration branchy wait selection).
// mode 0: normal epilogue (Cf f32 / Cb bf16)
// mode 1: qkv epilogue -> scatter to q,k (B,NH,S,HD) and vT (B,NH,HD,S)
// Optional per-batch (blockIdx.z): A/C strided, Bt/bias indexed by gidx.
// ---------------------------------------------------------------------------
#define KSTEP 32
#define LDSP  40   // padded row length (bf16 elems)

__global__ __launch_bounds__(256) void gemm_bf16_k(
    const bf16_t* __restrict__ A, const bf16_t* __restrict__ Bt,
    const float* __restrict__ bias, float* __restrict__ Cf, bf16_t* __restrict__ Cb,
    int M, int N, int K, const int* __restrict__ gidx,
    long long aBatch, long long cBatch, int relu,
    bf16_t* __restrict__ oq, bf16_t* __restrict__ ok, bf16_t* __restrict__ ov,
    int mode)
{
  __shared__ bf16_t sA[2][128][LDSP];
  __shared__ bf16_t sB[2][128][LDSP];

  const int tid  = threadIdx.x;
  const int lane = tid & 31;
  const int w    = tid >> 5;
  const int wm   = w & 3;        // 4 waves along M
  const int wn   = w >> 2;       // 2 waves along N
  const int mBase = blockIdx.y * 128;
  const int nBase = blockIdx.x * 128;
  const int z     = blockIdx.z;

  if (gidx) {
    int g = gidx[z];
    Bt += (size_t)g * N * K;
    if (bias) bias += (size_t)g * N;
  }
  A += (size_t)z * aBatch;
  if (Cf) Cf += (size_t)z * cBatch;
  if (Cb) Cb += (size_t)z * cBatch;

  const int r  = lane & 15;
  const int hf = lane >> 4;

  // cooperative staging: thread -> one 32B (16-elem) chunk of A and B per tile
  const int cRow  = tid >> 1;          // 0..127
  const int cHalf = (tid & 1) * 16;    // elem offset 0 or 16
  const bf16_t* gA = A  + (size_t)(mBase + cRow) * K + cHalf;
  const bf16_t* gB = Bt + (size_t)(nBase + cRow) * K + cHalf;

  auto stage = [&](int st, int k0) {
    copy_b128(gA + k0,     &sA[st][cRow][cHalf]);
    copy_b128(gA + k0 + 8, &sA[st][cRow][cHalf + 8]);
    copy_b128(gB + k0,     &sB[st][cRow][cHalf]);
    copy_b128(gB + k0 + 8, &sB[st][cRow][cHalf + 8]);
  };

  f32x8 acc[2][4];
#pragma unroll
  for (int i = 0; i < 2; ++i)
#pragma unroll
    for (int j = 0; j < 4; ++j) acc[i][j] = zero8();

  auto compute = [&](int st) {
    BF16Frag af[2], bfrag[4];
#pragma unroll
    for (int mi = 0; mi < 2; ++mi) {
      const int arow = wm * 32 + mi * 16 + r;
      af[mi].h[0] = *(const bf16x8*)&sA[st][arow][hf * 8];
      af[mi].h[1] = *(const bf16x8*)&sA[st][arow][16 + hf * 8];
    }
#pragma unroll
    for (int ni = 0; ni < 4; ++ni) {
      const int brow = wn * 64 + ni * 16 + r;
      bfrag[ni].h[0] = *(const bf16x8*)&sB[st][brow][hf * 8];
      bfrag[ni].h[1] = *(const bf16x8*)&sB[st][brow][16 + hf * 8];
    }
#pragma unroll
    for (int mi = 0; mi < 2; ++mi)
#pragma unroll
      for (int ni = 0; ni < 4; ++ni)
        acc[mi][ni] = __builtin_amdgcn_wmma_f32_16x16x32_bf16(
            false, af[mi].v, false, bfrag[ni].v, (short)0, acc[mi][ni], false, false);
  };

  // 2-deep pipeline: preload stages 0 and 1
  stage(0, 0);
  if (K > KSTEP) stage(1, KSTEP);

  // steady state: one newer group always in flight -> constant wait count
  int st = 0;
  int k0 = 0;
  for (; k0 + KSTEP < K; k0 += KSTEP) {
    wait_async_4();                    // current stage landed
    __syncthreads();
    compute(st);
    __syncthreads();                   // everyone done reading stage st
    if (k0 + 2 * KSTEP < K) stage(st, k0 + 2 * KSTEP);
    st ^= 1;
  }
  // peeled final k-step: drain everything, no trailing barrier/stage
  wait_async_0();
  __syncthreads();
  compute(st);

  // epilogue; C layout: col = lane%16, row = 8*(lane/16)+v
  const int m0 = mBase + wm * 32;
  const int n0 = nBase + wn * 64;
#pragma unroll
  for (int mi = 0; mi < 2; ++mi) {
#pragma unroll
    for (int ni = 0; ni < 4; ++ni) {
      const int col = n0 + ni * 16 + r;
      const float bv = bias ? bias[col] : 0.0f;
#pragma unroll
      for (int v = 0; v < 8; ++v) {
        const int row = m0 + mi * 16 + hf * 8 + v;
        float val = acc[mi][ni][v] + bv;
        if (relu) val = fmaxf(val, 0.0f);
        if (mode == 1) {
          // row = b*S+s over BS; col in [0,3H): sect | nh | hd
          const int sect = col >> 9;
          const int h    = col & (H_ - 1);
          const int nh   = h >> 6;
          const int hd   = h & 63;
          const int bb   = row >> 9;
          const int ss   = row & (S_ - 1);
          const bf16_t bvv = f2bf(val);
          if (sect == 0)
            oq[(((size_t)bb * NH_ + nh) * S_ + ss) * HD_ + hd] = bvv;
          else if (sect == 1)
            ok[(((size_t)bb * NH_ + nh) * S_ + ss) * HD_ + hd] = bvv;
          else
            ov[(((size_t)bb * NH_ + nh) * HD_ + hd) * S_ + ss] = bvv;
        } else {
          const size_t idx = (size_t)row * N + col;
          if (Cf) Cf[idx] = val;
          if (Cb) Cb[idx] = f2bf(val);
        }
      }
    }
  }
}

// ---------------------------------------------------------------------------
// Attention: per (b, head), each wave owns a 16-row Q tile.
// scores (16x512) in a per-wave LDS slice; wave32 shfl_xor softmax; then
// P(bf16) @ V^T with WMMA. Block = 8 waves -> 128 q rows; LDS = 256KB
// (within CDNA5's 320KB/WGP workgroup limit).
// ---------------------------------------------------------------------------
__global__ __launch_bounds__(256) void attn_k(
    const bf16_t* __restrict__ q, const bf16_t* __restrict__ k,
    const bf16_t* __restrict__ vT, const int* __restrict__ mask,
    bf16_t* __restrict__ out, float scale)
{
  extern __shared__ float smem[];
  const int lane = threadIdx.x & 31;
  const int w    = threadIdx.x >> 5;
  const int b    = blockIdx.z;
  const int nh   = blockIdx.y;
  const int q0   = blockIdx.x * 128 + w * 16;
  float* sc = smem + (size_t)w * 16 * S_;   // [16][512] per wave

  const int r  = lane & 15;
  const int hf = lane >> 4;

  const bf16_t* qp = q  + (((size_t)b * NH_ + nh) * S_ + q0) * HD_;
  const bf16_t* kp = k  + ((size_t)b * NH_ + nh) * S_ * HD_;
  const bf16_t* vp = vT + ((size_t)b * NH_ + nh) * HD_ * S_;
  const int*    mp = mask + (size_t)b * S_;

  // Q A-frags (K = HD = 64 -> two 16x16x32 steps)
  BF16Frag qa[2];
#pragma unroll
  for (int kk = 0; kk < 2; ++kk) {
    const bf16_t* p = qp + (size_t)r * HD_ + kk * 32 + hf * 8;
    qa[kk].h[0] = *(const bf16x8*)p;
    qa[kk].h[1] = *(const bf16x8*)(p + 16);
  }

  // scores = Q K^T * scale + mask-bias -> LDS
  for (int j = 0; j < S_ / 16; ++j) {
    BF16Frag kb0, kb1;
    const bf16_t* p = kp + (size_t)(j * 16 + r) * HD_ + hf * 8;
    kb0.h[0] = *(const bf16x8*)p;
    kb0.h[1] = *(const bf16x8*)(p + 16);
    kb1.h[0] = *(const bf16x8*)(p + 32);
    kb1.h[1] = *(const bf16x8*)(p + 48);
    f32x8 c = zero8();
    c = __builtin_amdgcn_wmma_f32_16x16x32_bf16(false, qa[0].v, false, kb0.v, (short)0, c, false, false);
    c = __builtin_amdgcn_wmma_f32_16x16x32_bf16(false, qa[1].v, false, kb1.v, (short)0, c, false, false);
    const int   col = j * 16 + r;
    const float mb  = (mp[col] == 0) ? -1.0e9f : 0.0f;
#pragma unroll
    for (int v = 0; v < 8; ++v)
      sc[(size_t)(hf * 8 + v) * S_ + col] = c[v] * scale + mb;
  }

  // softmax: lanes (r, r+16) handle row r, 256 cols each; combine via xor-16
  float* rowp = sc + (size_t)r * S_ + hf * 256;
  float mx = -3.0e38f;
  for (int i = 0; i < 256; ++i) mx = fmaxf(mx, rowp[i]);
  mx = fmaxf(mx, __shfl_xor(mx, 16, 32));
  float sum = 0.0f;
  for (int i = 0; i < 256; ++i) {
    float e = __expf(rowp[i] - mx);
    rowp[i] = e;
    sum += e;
  }
  sum += __shfl_xor(sum, 16, 32);
  const float inv = 1.0f / sum;

  // attn = P @ V  (K = S = 512, N = HD = 64 -> 4 C-frags)
  f32x8 o[4];
#pragma unroll
  for (int n = 0; n < 4; ++n) o[n] = zero8();
  const float* prow = sc + (size_t)r * S_;
  for (int k0 = 0; k0 < S_; k0 += 32) {
    BF16Frag pa;
#pragma unroll
    for (int i = 0; i < 8; ++i) {
      pa.h[0][i] = f2bf(prow[k0 + hf * 16 + i] * inv);
      pa.h[1][i] = f2bf(prow[k0 + hf * 16 + 8 + i] * inv);
    }
#pragma unroll
    for (int n = 0; n < 4; ++n) {
      BF16Frag vb;
      vb.v = *(const bf16x16*)(vp + (size_t)(n * 16 + r) * S_ + k0 + hf * 16);
      o[n] = __builtin_amdgcn_wmma_f32_16x16x32_bf16(
          false, pa.v, false, vb.v, (short)0, o[n], false, false);
    }
  }

#pragma unroll
  for (int n = 0; n < 4; ++n)
#pragma unroll
    for (int v = 0; v < 8; ++v) {
      const int row = q0 + hf * 8 + v;
      out[((size_t)b * S_ + row) * H_ + nh * HD_ + n * 16 + r] = f2bf(o[n][v]);
    }
}

// ---------------------------------------------------------------------------
// x = LN(resid + delta) * g + b   (wave per row; wave32 shfl reductions)
// writes both f32 (residual stream) and bf16 (next GEMM A matrix)
// ---------------------------------------------------------------------------
__global__ __launch_bounds__(256) void ln_residual_k(
    const float* __restrict__ resid, const float* __restrict__ delta,
    const float* __restrict__ gw, const float* __restrict__ bw,
    float* __restrict__ xout, bf16_t* __restrict__ xbf)
{
  const int lane = threadIdx.x & 31;
  const int w    = threadIdx.x >> 5;
  const size_t row = (size_t)blockIdx.x * 8 + w;
  const float* rp = resid + row * H_;
  const float* dp = delta + row * H_;
  float vals[16];
  float s = 0.0f;
#pragma unroll
  for (int i = 0; i < 16; ++i) {
    float t = rp[lane + i * 32] + dp[lane + i * 32];
    vals[i] = t;
    s += t;
  }
#pragma unroll
  for (int off = 16; off >= 1; off >>= 1) s += __shfl_xor(s, off, 32);
  const float mean = s * (1.0f / H_);
  float vs = 0.0f;
#pragma unroll
  for (int i = 0; i < 16; ++i) { float d = vals[i] - mean; vs += d * d; }
#pragma unroll
  for (int off = 16; off >= 1; off >>= 1) vs += __shfl_xor(vs, off, 32);
  const float rstd = rsqrtf(vs * (1.0f / H_) + 1e-5f);
#pragma unroll
  for (int i = 0; i < 16; ++i) {
    const int h = lane + i * 32;
    const float o = (vals[i] - mean) * rstd * gw[h] + bw[h];
    xout[row * H_ + h] = o;
    xbf[row * H_ + h]  = f2bf(o);
  }
}

// ---------------------------------------------------------------------------
// x = relu(LN(se; obs_ln[g])) + act_emb[g,a] + rtg*ret_W + ret_b
//     + time_emb[t] + game_emb[g]
// ---------------------------------------------------------------------------
__global__ __launch_bounds__(256) void embed_k(
    const float* __restrict__ se, const int* __restrict__ gi,
    const int* __restrict__ actions, const float* __restrict__ rtg,
    const int* __restrict__ tsteps, const float* __restrict__ game_emb,
    const float* __restrict__ oln_g, const float* __restrict__ oln_b,
    const float* __restrict__ act_emb, const float* __restrict__ ret_W,
    const float* __restrict__ ret_b, const float* __restrict__ time_emb,
    float* __restrict__ xout, bf16_t* __restrict__ xbf)
{
  const int lane = threadIdx.x & 31;
  const int w    = threadIdx.x >> 5;
  const size_t row = (size_t)blockIdx.x * 8 + w;
  const int b = (int)(row >> 9);
  const int g = gi[b];
  const float* sp = se + row * H_;

  float vals[16];
  float s = 0.0f;
#pragma unroll
  for (int i = 0; i < 16; ++i) { float t = sp[lane + i * 32]; vals[i] = t; s += t; }
#pragma unroll
  for (int off = 16; off >= 1; off >>= 1) s += __shfl_xor(s, off, 32);
  const float mean = s * (1.0f / H_);
  float vs = 0.0f;
#pragma unroll
  for (int i = 0; i < 16; ++i) { float d = vals[i] - mean; vs += d * d; }
#pragma unroll
  for (int off = 16; off >= 1; off >>= 1) vs += __shfl_xor(vs, off, 32);
  const float rstd = rsqrtf(vs * (1.0f / H_) + 1e-5f);

  const int   a  = actions[row];
  const int   ts = tsteps[row];
  const float rv = rtg[row];
#pragma unroll
  for (int i = 0; i < 16; ++i) {
    const int h = lane + i * 32;
    float ln = (vals[i] - mean) * rstd * oln_g[(size_t)g * H_ + h] + oln_b[(size_t)g * H_ + h];
    ln = fmaxf(ln, 0.0f);
    const float o = ln
        + act_emb[((size_t)g * A_ + a) * H_ + h]
        + rv * ret_W[h] + ret_b[h]
        + time_emb[(size_t)ts * H_ + h]
        + game_emb[(size_t)g * H_ + h];
    xout[row * H_ + h] = o;
    xbf[row * H_ + h]  = f2bf(o);
  }
}

// ---------------------------------------------------------------------------
// action head (N = 18, too narrow for WMMA tiles): VALU dot products
// ---------------------------------------------------------------------------
__global__ void action_head_k(const float* __restrict__ x, const int* __restrict__ gi,
                              const float* __restrict__ ahW, const float* __restrict__ ahb,
                              float* __restrict__ out)
{
  const int row = blockIdx.x;
  const int o   = threadIdx.x;
  if (o >= A_) return;
  const int g = gi[row >> 9];
  const float* xr = x + (size_t)row * H_;
  const float* wp = ahW + (size_t)g * H_ * A_ + o;
  float acc = ahb[(size_t)g * A_ + o];
  for (int h = 0; h < H_; ++h) acc = fmaf(xr[h], wp[(size_t)h * A_], acc);
  out[(size_t)row * A_ + o] = acc;
}

__global__ __launch_bounds__(256) void return_head_k(
    const float* __restrict__ x, const float* __restrict__ rhW,
    const float* __restrict__ rhb, float* __restrict__ out)
{
  const int lane = threadIdx.x & 31;
  const int w    = threadIdx.x >> 5;
  const size_t row = (size_t)blockIdx.x * 8 + w;
  const float* xr = x + row * H_;
  float acc = 0.0f;
#pragma unroll
  for (int i = 0; i < 16; ++i) acc = fmaf(xr[lane + i * 32], rhW[lane + i * 32], acc);
#pragma unroll
  for (int off = 16; off >= 1; off >>= 1) acc += __shfl_xor(acc, off, 32);
  if (lane == 0) out[row] = acc + rhb[0];
}

// ---------------------------------------------------------------------------
// host driver
// ---------------------------------------------------------------------------
extern "C" void kernel_launch(void* const* d_in, const int* in_sizes, int n_in,
                              void* d_out, int out_size, void* d_ws, size_t ws_size,
                              hipStream_t stream)
{
  (void)in_sizes; (void)n_in; (void)out_size; (void)ws_size;

  const int*   gi       = (const int*)  d_in[0];
  const float* states   = (const float*)d_in[1];
  const int*   actions  = (const int*)  d_in[2];
  const float* rtg      = (const float*)d_in[3];
  const int*   tsteps   = (const int*)  d_in[4];
  const int*   amask    = (const int*)  d_in[5];
  const float* game_emb = (const float*)d_in[6];
  const float* obs_W    = (const float*)d_in[7];
  const float* obs_b    = (const float*)d_in[8];
  const float* obs_ln_g = (const float*)d_in[9];
  const float* obs_ln_b = (const float*)d_in[10];
  const float* act_emb  = (const float*)d_in[11];
  const float* ret_W    = (const float*)d_in[12];
  const float* ret_b    = (const float*)d_in[13];
  const float* time_emb = (const float*)d_in[14];
  const float* Wqkv     = (const float*)d_in[15];
  const float* bqkv     = (const float*)d_in[16];
  const float* Wo       = (const float*)d_in[17];
  const float* bo       = (const float*)d_in[18];
  const float* ln1_g    = (const float*)d_in[19];
  const float* ln1_b    = (const float*)d_in[20];
  const float* W1       = (const float*)d_in[21];
  const float* b1       = (const float*)d_in[22];
  const float* W2       = (const float*)d_in[23];
  const float* b2       = (const float*)d_in[24];
  const float* ln2_g    = (const float*)d_in[25];
  const float* ln2_b    = (const float*)d_in[26];
  const float* sh_W     = (const float*)d_in[27];
  const float* sh_b     = (const float*)d_in[28];
  const float* ah_W     = (const float*)d_in[29];
  const float* ah_b     = (const float*)d_in[30];
  const float* rh_W     = (const float*)d_in[31];
  const float* rh_b     = (const float*)d_in[32];

  char* ws = (char*)d_ws;
  size_t off = 0;
  auto take = [&](size_t bytes) -> char* {
    char* p = ws + off;
    off += (bytes + 255) & ~(size_t)255;
    return p;
  };

  bf16_t* obsWt_bf  = (bf16_t*)take((size_t)G_ * H_ * D_ * 2);
  bf16_t* Wqkv_bf   = (bf16_t*)take((size_t)NL_ * 3 * H_ * H_ * 2);
  bf16_t* Wo_bf     = (bf16_t*)take((size_t)NL_ * H_ * H_ * 2);
  bf16_t* W1_bf     = (bf16_t*)take((size_t)NL_ * 4 * H_ * H_ * 2);
  bf16_t* W2_bf     = (bf16_t*)take((size_t)NL_ * H_ * 4 * H_ * 2);
  bf16_t* shWt_bf   = (bf16_t*)take((size_t)G_ * D_ * H_ * 2);
  bf16_t* states_bf = (bf16_t*)take((size_t)BS_ * D_ * 2);
  bf16_t* x_bf      = (bf16_t*)take((size_t)BS_ * H_ * 2);
  bf16_t* big_bf    = (bf16_t*)take((size_t)BS_ * 4 * H_ * 2);   // ff intermediate
  bf16_t* q_bf      = (bf16_t*)take((size_t)BS_ * H_ * 2);
  bf16_t* k_bf      = (bf16_t*)take((size_t)BS_ * H_ * 2);
  bf16_t* vT_bf     = (bf16_t*)take((size_t)BS_ * H_ * 2);
  bf16_t* attn_bf   = (bf16_t*)take((size_t)BS_ * H_ * 2);
  float*  x_f32     = (float*) take((size_t)BS_ * H_ * 4);
  float*  y_f32     = (float*) take((size_t)BS_ * H_ * 4);

  const dim3 blk(256);

  // weight / input conversion (bf16)
  cvt_bf16_k<<<4096, 256, 0, stream>>>(Wqkv, Wqkv_bf, (size_t)NL_ * 3 * H_ * H_);
  cvt_bf16_k<<<4096, 256, 0, stream>>>(Wo,   Wo_bf,   (size_t)NL_ * H_ * H_);
  cvt_bf16_k<<<4096, 256, 0, stream>>>(W1,   W1_bf,   (size_t)NL_ * 4 * H_ * H_);
  cvt_bf16_k<<<4096, 256, 0, stream>>>(W2,   W2_bf,   (size_t)NL_ * H_ * 4 * H_);
  cvt_bf16_k<<<4096, 256, 0, stream>>>(states, states_bf, (size_t)BS_ * D_);
  cvt_bf16_t_k<<<2048, 256, 0, stream>>>(obs_W, obsWt_bf, G_, D_, H_); // (D,H)->(H,D)
  cvt_bf16_t_k<<<2048, 256, 0, stream>>>(sh_W,  shWt_bf,  G_, H_, D_); // (H,D)->(D,H)

  // se = states @ obs_W[g] + obs_b[g]   (per-batch game weights)
  gemm_bf16_k<<<dim3(H_ / 128, S_ / 128, B_), blk, 0, stream>>>(
      states_bf, obsWt_bf, obs_b, y_f32, nullptr,
      S_, H_, D_, gi, (long long)S_ * D_, (long long)S_ * H_, 0,
      nullptr, nullptr, nullptr, 0);

  // x = relu(LN(se)) + ae + re + te + g_emb
  embed_k<<<BS_ / 8, blk, 0, stream>>>(
      y_f32, gi, actions, rtg, tsteps, game_emb, obs_ln_g, obs_ln_b,
      act_emb, ret_W, ret_b, time_emb, x_f32, x_bf);

  const float scale = 0.125f; // 1/sqrt(HD)
  for (int l = 0; l < NL_; ++l) {
    // qkv = x @ Wqkv^T + bqkv  -> fused scatter to q,k,vT (mode 1)
    gemm_bf16_k<<<dim3(3 * H_ / 128, BS_ / 128, 1), blk, 0, stream>>>(
        x_bf, Wqkv_bf + (size_t)l * 3 * H_ * H_, bqkv + (size_t)l * 3 * H_,
        nullptr, nullptr, BS_, 3 * H_, H_, nullptr, 0, 0, 0,
        q_bf, k_bf, vT_bf, 1);

    attn_k<<<dim3(S_ / 128, NH_, B_), blk, 8 * 16 * S_ * sizeof(float), stream>>>(
        q_bf, k_bf, vT_bf, amask, attn_bf, scale);

    // attn @ Wo^T + bo -> f32
    gemm_bf16_k<<<dim3(H_ / 128, BS_ / 128, 1), blk, 0, stream>>>(
        attn_bf, Wo_bf + (size_t)l * H_ * H_, bo + (size_t)l * H_,
        y_f32, nullptr, BS_, H_, H_, nullptr, 0, 0, 0,
        nullptr, nullptr, nullptr, 0);
    ln_residual_k<<<BS_ / 8, blk, 0, stream>>>(
        x_f32, y_f32, ln1_g + (size_t)l * H_, ln1_b + (size_t)l * H_, x_f32, x_bf);

    // ff = relu(x @ W1^T + b1) -> bf16
    gemm_bf16_k<<<dim3(4 * H_ / 128, BS_ / 128, 1), blk, 0, stream>>>(
        x_bf, W1_bf + (size_t)l * 4 * H_ * H_, b1 + (size_t)l * 4 * H_,
        nullptr, big_bf, BS_, 4 * H_, H_, nullptr, 0, 0, 1,
        nullptr, nullptr, nullptr, 0);
    // ff @ W2^T + b2 -> f32
    gemm_bf16_k<<<dim3(H_ / 128, BS_ / 128, 1), blk, 0, stream>>>(
        big_bf, W2_bf + (size_t)l * H_ * 4 * H_, b2 + (size_t)l * H_,
        y_f32, nullptr, BS_, H_, 4 * H_, nullptr, 0, 0, 0,
        nullptr, nullptr, nullptr, 0);
    ln_residual_k<<<BS_ / 8, blk, 0, stream>>>(
        x_f32, y_f32, ln2_g + (size_t)l * H_, ln2_b + (size_t)l * H_, x_f32, x_bf);
  }

  float* out_state  = (float*)d_out;                       // (B,S,D)
  float* out_action = out_state + (size_t)BS_ * D_;        // (B,S,A)
  float* out_return = out_action + (size_t)BS_ * A_;       // (B,S,1)

  // state head: x @ sh_W[g] + sh_b[g]   (per-batch game weights)
  gemm_bf16_k<<<dim3(D_ / 128, S_ / 128, B_), blk, 0, stream>>>(
      x_bf, shWt_bf, sh_b, out_state, nullptr,
      S_, D_, H_, gi, (long long)S_ * H_, (long long)S_ * D_, 0,
      nullptr, nullptr, nullptr, 0);

  action_head_k<<<BS_, 32, 0, stream>>>(x_f32, gi, ah_W, ah_b, out_action);
  return_head_k<<<BS_ / 8, blk, 0, stream>>>(x_f32, rh_W, rh_b, out_return);
}
